// decoder_2525440770176
// MI455X (gfx1250) — matrix operand order repaired and checked by reading, not compiled, not run on hardware
//
#include <hip/hip_runtime.h>
#include <hip/hip_bf16.h>
#include <math.h>

// ============================================================================
// MoE-ViT decoder for MI455X (gfx1250, wave32).
//  - Large GEMMs (qkv, w_o, expert FFN) on v_wmma_f32_16x16x32_f16.
//    One wave computes a 32x32 output tile (2x2 fragments, 4 WMMA per K-step).
//  - Weights pre-packed to f16 in WMMA fragment order -> B loads are
//    contiguous b128s; activations loaded as float4 + v_cvt_pk_f16_f32.
//  - __builtin_prefetch on next K-tile of packed B (global_prefetch_b8).
//  - Top-2 expert skip: per 32-row tile, wave-uniform __any(gate>0) early-out
//    (mathematically exact: gate==0 experts contribute nothing).
//  - Channel-attention gram (dh down to 8) + softmax + attn*v: scalar
//    reduction kernels (<2% of FLOPs).
//  - Upsample = fused conv3x3 + PixelShuffle(2) + skip add, token-major out.
//  - Working set (~90 MB incl. packed weights) fits in 192 MB L2; HBM traffic
//    (~80 MB total) is negligible at 23.3 TB/s -> compute-bound on WMMA.
// ============================================================================

typedef __attribute__((ext_vector_type(16))) _Float16 v16h;
typedef __attribute__((ext_vector_type(8)))  float    v8f;

#define HEADS_ 4
#define NEXP_  8

// ---------------- workspace layout (float offsets) ----------------
static constexpr size_t OFF_TA    = 0;          // 2,097,152 floats
static constexpr size_t OFF_TB    = 2097152;    // 2,097,152
static constexpr size_t OFF_W1    = 4194304;    // 6,291,456 (qkv / h2)
static constexpr size_t OFF_W2    = 10485760;   // 8,388,608 (h / o / hmid)
static constexpr size_t OFF_GATE  = 18874368;   // 524,288
static constexpr size_t OFF_SMALL = 19398656;   // norms/gram/attn/rbias/auxp
static constexpr size_t OFF_PACK  = 19464192;   // 4,456,448 halves (2.23M floats)

// ---------------- device helpers ----------------
__device__ __forceinline__ float gelu_tanh(float x) {
  float x3 = x * x * x;
  return 0.5f * x * (1.0f + tanhf(0.7978845608028654f * (x + 0.044715f * x3)));
}

// Load one 16x32 A fragment (f32 row-major activations -> f16 fragment).
// Lane layout per ISA: idx 0-7 -> K = hl*8 + 0..7 ; idx 8-15 -> K+16.
__device__ __forceinline__ v16h load_a_frag(const float* __restrict__ p) {
  const float4* q = (const float4*)p;
  float4 x0 = q[0], x1 = q[1], x2 = q[4], x3 = q[5];
  v16h a;
  a[0]  = (_Float16)x0.x; a[1]  = (_Float16)x0.y;
  a[2]  = (_Float16)x0.z; a[3]  = (_Float16)x0.w;
  a[4]  = (_Float16)x1.x; a[5]  = (_Float16)x1.y;
  a[6]  = (_Float16)x1.z; a[7]  = (_Float16)x1.w;
  a[8]  = (_Float16)x2.x; a[9]  = (_Float16)x2.y;
  a[10] = (_Float16)x2.z; a[11] = (_Float16)x2.w;
  a[12] = (_Float16)x3.x; a[13] = (_Float16)x3.y;
  a[14] = (_Float16)x3.z; a[15] = (_Float16)x3.w;
  return a;
}

#define EPI_NONE      0
#define EPI_ADD       1
#define EPI_BIAS_GELU 2
#define EPI_GATE_ADD  3

// ---------------- WMMA GEMM: Out(M x Nc) = A(M x K) @ B(K x Nc) -------------
// A: f32 row-major activations.  Bp: f16 weights pre-packed in fragment order
// [ktile][ctile][lane][16].  grid = (M/32, Nc/32); block = 32 (one wave).
// M, Nc multiples of 32; K multiple of 32.
template<int EPI, int GATED>
__global__ __launch_bounds__(32)
void gemm32x32(const float* __restrict__ A, int lda,
               const _Float16* __restrict__ Bp, int ncTiles,
               float* __restrict__ Out, int ldo, int K,
               const float* __restrict__ bias,
               const float* __restrict__ gate, int eidx)
{
  const int lane = threadIdx.x;
  const int hl   = lane >> 4;
  const int l16  = lane & 15;
  const int row0 = blockIdx.x * 32;
  const int ct0  = blockIdx.y * 2;

  if (GATED) {
    // wave-uniform skip: 32 lanes cover exactly the 32 rows of this tile.
    float g = gate[(size_t)(row0 + lane) * NEXP_ + eidx];
    if (!__any(g > 0.0f)) return;
  }

  v8f acc00 = {}, acc01 = {}, acc10 = {}, acc11 = {};
  const int nKt = K >> 5;
  const float* arow0 = A + (size_t)(row0 + l16) * lda + (hl << 3);
  const float* arow1 = A + (size_t)(row0 + 16 + l16) * lda + (hl << 3);

  for (int kt = 0; kt < nKt; ++kt) {
    const size_t fbase = ((size_t)(kt * ncTiles + ct0) << 9) + (lane << 4);
    v16h b0 = *(const v16h*)(Bp + fbase);
    v16h b1 = *(const v16h*)(Bp + fbase + 512);
    v16h a0 = load_a_frag(arow0 + (kt << 5));
    v16h a1 = load_a_frag(arow1 + (kt << 5));
    if (kt + 1 < nKt)  // pull next packed-B panel toward the WGP (prefetch_b8)
      __builtin_prefetch(Bp + (((size_t)((kt + 1) * ncTiles + ct0)) << 9), 0, 0);
    acc00 = __builtin_amdgcn_wmma_f32_16x16x32_f16(false, a0, false, b0, (short)0, acc00, false, false);
    acc01 = __builtin_amdgcn_wmma_f32_16x16x32_f16(false, a0, false, b1, (short)0, acc01, false, false);
    acc10 = __builtin_amdgcn_wmma_f32_16x16x32_f16(false, a1, false, b0, (short)0, acc10, false, false);
    acc11 = __builtin_amdgcn_wmma_f32_16x16x32_f16(false, a1, false, b1, (short)0, acc11, false, false);
  }

#pragma unroll
  for (int mt = 0; mt < 2; ++mt) {
#pragma unroll
    for (int nt = 0; nt < 2; ++nt) {
      v8f acc = (mt == 0) ? (nt == 0 ? acc00 : acc01)
                          : (nt == 0 ? acc10 : acc11);
      int col = (ct0 + nt) * 16 + l16;
#pragma unroll
      for (int r = 0; r < 8; ++r) {
        int rowm = row0 + mt * 16 + r + (hl << 3);  // C/D layout
        size_t oi = (size_t)rowm * ldo + col;
        float v = acc[r];
        if (EPI == EPI_NONE) {
          Out[oi] = v;
        } else if (EPI == EPI_ADD) {
          Out[oi] += v;
        } else if (EPI == EPI_BIAS_GELU) {
          Out[oi] = gelu_tanh(v + bias[col]);
        } else {  // EPI_GATE_ADD
          float g = gate[(size_t)rowm * NEXP_ + eidx];
          Out[oi] += g * (v + bias[col]);
        }
      }
    }
  }
}

// Pack f32 weight matrix W[K x Nc] (row-major) into f16 WMMA B-fragment order:
// dst[((kt*ncTiles + ct)*512) + lane*16 + idx] = W[kt*32 + 8*(lane>>4) + (idx&7) + 16*(idx>>3),
//                                                  ct*16 + (lane&15)].
// grid.z batches experts (stride K*Nc on both sides).
__global__ void pack_b_kernel(const float* __restrict__ W,
                              _Float16* __restrict__ P, int K, int Nc) {
  size_t base = (size_t)blockIdx.z * K * Nc;
  long idx = (long)blockIdx.x * blockDim.x + threadIdx.x;
  if (idx >= (long)K * Nc) return;
  int id16 = (int)(idx & 15);
  int lane = (int)((idx >> 4) & 31);
  long frag = idx >> 9;
  int nct = Nc >> 4;
  int ct = (int)(frag % nct);
  int kt = (int)(frag / nct);
  int k = kt * 32 + ((lane >> 4) << 3) + (id16 & 7) + ((id16 >> 3) << 4);
  int c = ct * 16 + (lane & 15);
  P[base + idx] = (_Float16)W[base + (size_t)k * Nc + c];
}

// ---------------- misc kernels ----------------
__global__ void zero_kernel(float* p, int n) {
  int i = blockIdx.x * blockDim.x + threadIdx.x;
  if (i < n) p[i] = 0.0f;
}

// x [C,N] image-major -> t [N,C] token-major
__global__ void transpose_cn_kernel(const float* __restrict__ x,
                                    float* __restrict__ t, int C, int N) {
  long idx = (long)blockIdx.x * blockDim.x + threadIdx.x;
  if (idx >= (long)C * N) return;
  int c = (int)(idx / N);
  int n = (int)(idx % N);
  t[(size_t)n * C + c] = x[idx];
}

// LayerNorm over channel dim per token: h = (t-m)*rsqrt(v+1e-5)*g + b
__global__ void ln_kernel(const float* __restrict__ t, float* __restrict__ h,
                          const float* __restrict__ g, const float* __restrict__ b,
                          int N, int C) {
  int n = blockIdx.x * blockDim.x + threadIdx.x;
  if (n >= N) return;
  const float* tr = t + (size_t)n * C;
  float m = 0.0f, s2 = 0.0f;
  for (int c = 0; c < C; ++c) { float v = tr[c]; m += v; s2 += v * v; }
  m /= C;
  float var = s2 / C - m * m;
  float r = rsqrtf(var + 1e-5f);
  float* hr = h + (size_t)n * C;
  for (int c = 0; c < C; ++c) hr[c] = (tr[c] - m) * r * g[c] + b[c];
}

// L2 norm over tokens of column c (q,k columns of qkv). grid = 2C, block = 256.
__global__ void colnorm_kernel(const float* __restrict__ qkv,
                               float* __restrict__ norms, int N, int ld) {
  int c = blockIdx.x;
  __shared__ float red[256];
  float s = 0.0f;
  for (int n = threadIdx.x; n < N; n += 256) {
    float v = qkv[(size_t)n * ld + c];
    s += v * v;
  }
  red[threadIdx.x] = s; __syncthreads();
  for (int st = 128; st > 0; st >>= 1) {
    if (threadIdx.x < st) red[threadIdx.x] += red[threadIdx.x + st];
    __syncthreads();
  }
  if (threadIdx.x == 0) norms[c] = sqrtf(red[0]);
}

// qkv[:, 0:2C] /= (norm + 1e-6) in place
__global__ void colscale_kernel(float* __restrict__ qkv,
                                const float* __restrict__ norms,
                                long total, int twoC, int ld) {
  long idx = (long)blockIdx.x * blockDim.x + threadIdx.x;
  if (idx >= total) return;
  int c = (int)(idx % twoC);
  long n = idx / twoC;
  qkv[(size_t)n * ld + c] /= (norms[c] + 1e-6f);
}

// gram[h,d,e] = sum_n qn[n,h*dh+d] * kn[n,C+h*dh+e].  grid = HEADS*dh*dh.
__global__ void gram_kernel(const float* __restrict__ qkv,
                            float* __restrict__ gram, int N, int C, int dh) {
  int o = blockIdx.x;
  int e = o % dh; int t2 = o / dh; int d = t2 % dh; int hh = t2 / dh;
  const float* qc = qkv + hh * dh + d;
  const float* kc = qkv + C + hh * dh + e;
  int ld = 3 * C;
  __shared__ float red[256];
  float s = 0.0f;
  for (int n = threadIdx.x; n < N; n += 256)
    s += qc[(size_t)n * ld] * kc[(size_t)n * ld];
  red[threadIdx.x] = s; __syncthreads();
  for (int st = 128; st > 0; st >>= 1) {
    if (threadIdx.x < st) red[threadIdx.x] += red[threadIdx.x + st];
    __syncthreads();
  }
  if (threadIdx.x == 0) gram[o] = red[0];
}

// softmax over e of gram[h,d,:]*temp[h].  grid = HEADS*dh, block = dh (<=64)
__global__ void attn_softmax_kernel(const float* __restrict__ gram,
                                    float* __restrict__ attn,
                                    const float* __restrict__ temp, int dh) {
  int row = blockIdx.x;        // h*dh + d
  int hh = row / dh;
  int e = threadIdx.x;
  __shared__ float buf[64];
  float v = gram[(size_t)row * dh + e] * temp[hh];
  buf[e] = v; __syncthreads();
  for (int st = dh >> 1; st > 0; st >>= 1) {
    if (e < st) buf[e] = fmaxf(buf[e], buf[e + st]);
    __syncthreads();
  }
  float mx = buf[0]; __syncthreads();
  float ex = expf(v - mx);
  buf[e] = ex; __syncthreads();
  for (int st = dh >> 1; st > 0; st >>= 1) {
    if (e < st) buf[e] += buf[e + st];
    __syncthreads();
  }
  attn[(size_t)row * dh + e] = ex / buf[0];
}

// o[n, h*dh+d] = sum_e attn[h,d,e] * v[n, 2C + h*dh + e]
__global__ void attnv_kernel(const float* __restrict__ qkv,
                             const float* __restrict__ attn,
                             float* __restrict__ o, int N, int C, int dh) {
  long idx = (long)blockIdx.x * blockDim.x + threadIdx.x;
  if (idx >= (long)N * C) return;
  int c = (int)(idx % C);
  long n = idx / C;
  int hh = c / dh, d = c % dh;
  const float* ar = attn + (size_t)(hh * dh + d) * dh;
  const float* vr = qkv + (size_t)n * 3 * C + 2 * C + hh * dh;
  float s = 0.0f;
  for (int e = 0; e < dh; ++e) s += ar[e] * vr[e];
  o[idx] = s;
}

// rbias[e] = (route @ w_r)[e] + task_emb[task_id, e]
__global__ void rbias_kernel(const float* __restrict__ route,
                             const float* __restrict__ wr,
                             const float* __restrict__ temb,
                             const int* __restrict__ task,
                             float* __restrict__ rbias) {
  int e = threadIdx.x;
  if (e >= NEXP_) return;
  float s = temb[task[0] * NEXP_ + e];
  for (int j = 0; j < 512; ++j) s += route[j] * wr[j * NEXP_ + e];
  rbias[e] = s;
}

// per-token routing: softmax over 8, top-2 gates, aux-loss partials.
__global__ void routing_kernel(const float* __restrict__ h2,
                               const float* __restrict__ wg,
                               const float* __restrict__ rbias,
                               float* __restrict__ gate,
                               float* __restrict__ auxp, int N, int C) {
  __shared__ float sp[17];
  if (threadIdx.x < 17) sp[threadIdx.x] = 0.0f;
  __syncthreads();
  int n = blockIdx.x * blockDim.x + threadIdx.x;
  if (n < N) {
    float lg[NEXP_];
#pragma unroll
    for (int e = 0; e < NEXP_; ++e) lg[e] = rbias[e];
    const float* hr = h2 + (size_t)n * C;
    for (int c = 0; c < C; ++c) {
      float hv = hr[c];
      const float* wr = wg + (size_t)c * NEXP_;
#pragma unroll
      for (int e = 0; e < NEXP_; ++e) lg[e] += hv * wr[e];
    }
    float mx = lg[0];
#pragma unroll
    for (int e = 1; e < NEXP_; ++e) mx = fmaxf(mx, lg[e]);
    float p[NEXP_]; float s = 0.0f;
#pragma unroll
    for (int e = 0; e < NEXP_; ++e) { p[e] = expf(lg[e] - mx); s += p[e]; }
    float inv = 1.0f / s;
#pragma unroll
    for (int e = 0; e < NEXP_; ++e) p[e] *= inv;
    int i1 = 0;
#pragma unroll
    for (int e = 1; e < NEXP_; ++e) if (p[e] > p[i1]) i1 = e;
    int i2 = (i1 == 0) ? 1 : 0;
#pragma unroll
    for (int e = 0; e < NEXP_; ++e) if (e != i1 && p[e] > p[i2]) i2 = e;
    float gs = p[i1] + p[i2];
    float* gr = gate + (size_t)n * NEXP_;
#pragma unroll
    for (int e = 0; e < NEXP_; ++e) gr[e] = 0.0f;
    gr[i1] = p[i1] / gs;
    gr[i2] = p[i2] / gs;
    float ent = 0.0f;
#pragma unroll
    for (int e = 0; e < NEXP_; ++e) ent -= p[e] * logf(p[e] + 1e-6f);
#pragma unroll
    for (int e = 0; e < NEXP_; ++e) atomicAdd(&sp[e], p[e]);
    atomicAdd(&sp[8 + i1], 1.0f);
    atomicAdd(&sp[8 + i2], 1.0f);
    atomicAdd(&sp[16], ent);
  }
  __syncthreads();
  if (threadIdx.x < 17) atomicAdd(&auxp[threadIdx.x], sp[threadIdx.x]);
}

// fold this block's aux partials into the two output scalars
__global__ void aux_fin_kernel(const float* __restrict__ auxp,
                               float* __restrict__ outp, float invN) {
  if (threadIdx.x != 0 || blockIdx.x != 0) return;
  float stdl = 0.0f, entm = 0.0f;
  for (int e = 0; e < NEXP_; ++e) {
    float mp = auxp[e] * invN;
    float fr = auxp[8 + e] * invN;
    stdl += mp * fr;
    entm -= mp * logf(mp + 1e-6f);
  }
  stdl *= (float)NEXP_;
  float mi = auxp[16] * invN - entm;
  atomicAdd(&outp[0], stdl);
  atomicAdd(&outp[1], mi);
}

// conv3x3 (OIHW weights, SAME) + PixelShuffle(2) + skip add.
// input token-major [H*W, Cin]; skip image-major [Cout, 2H, 2W].
// TOKEN_OUT: write token-major [4HW, Cout]; else image-major [Cout, 2H, 2W].
template<int TOKEN_OUT>
__global__ void upsample_kernel(const float* __restrict__ tin,
                                const float* __restrict__ wt,
                                const float* __restrict__ skip,
                                float* __restrict__ out,
                                int Cin, int H, int W) {
  int Cout = Cin >> 1;
  int Wout = W * 2, Hout = H * 2;
  long total = (long)Hout * Wout * Cout;
  long idx = (long)blockIdx.x * blockDim.x + threadIdx.x;
  if (idx >= total) return;
  int c = (int)(idx % Cout);
  long np = idx / Cout;                 // output pixel (oh*Wout + ow)
  int ow = (int)(np % Wout);
  int oh = (int)(np / Wout);
  int h = oh >> 1, a = oh & 1, w = ow >> 1, b = ow & 1;
  int cc = c * 4 + a * 2 + b;           // pre-shuffle conv output channel
  float acc = skip[(size_t)c * Hout * Wout + (size_t)oh * Wout + ow];
  const float* wrow = wt + (size_t)cc * Cin * 9;
  for (int kh = 0; kh < 3; ++kh) {
    int ih = h + kh - 1;
    if (ih < 0 || ih >= H) continue;
    for (int kw = 0; kw < 3; ++kw) {
      int iw = w + kw - 1;
      if (iw < 0 || iw >= W) continue;
      const float* ip = tin + ((size_t)ih * W + iw) * Cin;
      const float* wp = wrow + kh * 3 + kw;
      for (int ci = 0; ci < Cin; ++ci) acc += ip[ci] * wp[(size_t)ci * 9];
    }
  }
  if (TOKEN_OUT) out[(size_t)np * Cout + c] = acc;
  else           out[(size_t)c * Hout * Wout + np] = acc;
}

// ---------------- host side ----------------
struct BlkP {
  const float *ln1_g, *ln1_b, *w_qkv, *w_o, *temp, *ln2_g, *ln2_b,
              *w_g, *w_r, *task_emb, *w1, *b1, *w2, *b2;
};

static void run_block(float* t, const BlkP& P, int C, int HH, int WW,
                      const float* route, const int* task,
                      float* dout_aux, float* wsf, hipStream_t stream) {
  const int N = HH * WW, hid = 4 * C, dh = C / HEADS_, C3 = 3 * C;
  float* work1 = wsf + OFF_W1;          // qkv, then h2
  float* work2 = wsf + OFF_W2;          // h, then o, then hmid
  float* gate  = wsf + OFF_GATE;
  float* norms = wsf + OFF_SMALL;       // 512
  float* gram  = norms + 512;           // 16384
  float* attn  = gram + 16384;          // 16384
  float* rbias = attn + 16384;          // 16
  float* auxp  = rbias + 16;            // 17
  _Float16* pk = (_Float16*)(wsf + OFF_PACK);
  _Float16* pk_qkv = pk;                                  // 3*C*C halves
  _Float16* pk_wo  = pk_qkv + (size_t)3 * C * C;          // C*C
  _Float16* pk_w1  = pk_wo + (size_t)C * C;               // 8*C*hid
  _Float16* pk_w2  = pk_w1 + (size_t)NEXP_ * C * hid;     // 8*hid*C

  // --- pack this block's weights into f16 WMMA fragment order ---
  pack_b_kernel<<<dim3((C * C3 + 255) / 256, 1, 1), 256, 0, stream>>>(
      P.w_qkv, pk_qkv, C, C3);
  pack_b_kernel<<<dim3((C * C + 255) / 256, 1, 1), 256, 0, stream>>>(
      P.w_o, pk_wo, C, C);
  pack_b_kernel<<<dim3((C * hid + 255) / 256, 1, NEXP_), 256, 0, stream>>>(
      P.w1, pk_w1, C, hid);
  pack_b_kernel<<<dim3((hid * C + 255) / 256, 1, NEXP_), 256, 0, stream>>>(
      P.w2, pk_w2, hid, C);

  // --- channel multi-head attention ---
  ln_kernel<<<(N + 255) / 256, 256, 0, stream>>>(t, work2, P.ln1_g, P.ln1_b, N, C);
  gemm32x32<EPI_NONE, 0><<<dim3(N / 32, C3 / 32), 32, 0, stream>>>(
      work2, C, pk_qkv, C3 / 16, work1, C3, C, nullptr, nullptr, 0);
  colnorm_kernel<<<2 * C, 256, 0, stream>>>(work1, norms, N, C3);
  long tot2 = (long)N * 2 * C;
  colscale_kernel<<<(int)((tot2 + 255) / 256), 256, 0, stream>>>(
      work1, norms, tot2, 2 * C, C3);
  gram_kernel<<<HEADS_ * dh * dh, 256, 0, stream>>>(work1, gram, N, C, dh);
  attn_softmax_kernel<<<HEADS_ * dh, dh, 0, stream>>>(gram, attn, P.temp, dh);
  long totNC = (long)N * C;
  attnv_kernel<<<(int)((totNC + 255) / 256), 256, 0, stream>>>(
      work1, attn, work2, N, C, dh);
  gemm32x32<EPI_ADD, 0><<<dim3(N / 32, C / 32), 32, 0, stream>>>(
      work2, C, pk_wo, C / 16, t, C, C, nullptr, nullptr, 0);

  // --- MoE FFN with top-2 routing ---
  ln_kernel<<<(N + 255) / 256, 256, 0, stream>>>(t, work1, P.ln2_g, P.ln2_b, N, C);
  zero_kernel<<<1, 32, 0, stream>>>(auxp, 17);
  rbias_kernel<<<1, 8, 0, stream>>>(route, P.w_r, P.task_emb, task, rbias);
  routing_kernel<<<(N + 127) / 128, 128, 0, stream>>>(
      work1, P.w_g, rbias, gate, auxp, N, C);
  for (int e = 0; e < NEXP_; ++e) {
    gemm32x32<EPI_BIAS_GELU, 1><<<dim3(N / 32, hid / 32), 32, 0, stream>>>(
        work1, C, pk_w1 + (size_t)e * C * hid, hid / 16, work2, hid, C,
        P.b1 + (size_t)e * hid, gate, e);
    gemm32x32<EPI_GATE_ADD, 1><<<dim3(N / 32, C / 32), 32, 0, stream>>>(
        work2, hid, pk_w2 + (size_t)e * hid * C, C / 16, t, C, hid,
        P.b2 + (size_t)e * C, gate, e);
  }
  aux_fin_kernel<<<1, 1, 0, stream>>>(auxp, dout_aux, 1.0f / (float)N);
}

extern "C" void kernel_launch(void* const* d_in, const int* in_sizes, int n_in,
                              void* d_out, int out_size, void* d_ws, size_t ws_size,
                              hipStream_t stream) {
  (void)n_in; (void)ws_size;
  float* wsf = (float*)d_ws;
  float* dout = (float*)d_out;

  // Input index maps: dict-insertion order vs pytree-alphabetical order.
  int IX[5], IROUTE, ITASK, IUP[4], IB[4][14];
  const bool insertion = (in_sizes[0] == 16 * 512 * 512);
  if (insertion) {
    for (int i = 0; i < 5; ++i) IX[i] = i;
    IROUTE = 5; ITASK = 6;
    // blk dict order: ln1_g,ln1_b,w_qkv,w_o,temp,ln2_g,ln2_b,w_g,w_r,task_emb,w1,b1,w2,b2
    for (int b = 0; b < 4; ++b)
      for (int j = 0; j < 14; ++j) IB[b][j] = 7 + b * 14 + j;
    for (int i = 0; i < 4; ++i) IUP[i] = 63 + i;
  } else {
    // alphabetical blk keys: b1,b2,ln1_b,ln1_g,ln2_b,ln2_g,task_emb,temp,w1,w2,w_g,w_o,w_qkv,w_r
    static const int perm[14] = {3, 2, 12, 11, 7, 5, 4, 10, 13, 6, 8, 0, 9, 1};
    for (int b = 0; b < 4; ++b)
      for (int j = 0; j < 14; ++j) IB[b][j] = b * 14 + perm[j];
    IROUTE = 56; ITASK = 57;
    for (int i = 0; i < 4; ++i) IUP[i] = 58 + i;
    IX[0] = 62; IX[1] = 63; IX[2] = 64; IX[3] = 65; IX[4] = 66;
  }

  BlkP blk[4];
  for (int b = 0; b < 4; ++b) {
    blk[b].ln1_g   = (const float*)d_in[IB[b][0]];
    blk[b].ln1_b   = (const float*)d_in[IB[b][1]];
    blk[b].w_qkv   = (const float*)d_in[IB[b][2]];
    blk[b].w_o     = (const float*)d_in[IB[b][3]];
    blk[b].temp    = (const float*)d_in[IB[b][4]];
    blk[b].ln2_g   = (const float*)d_in[IB[b][5]];
    blk[b].ln2_b   = (const float*)d_in[IB[b][6]];
    blk[b].w_g     = (const float*)d_in[IB[b][7]];
    blk[b].w_r     = (const float*)d_in[IB[b][8]];
    blk[b].task_emb= (const float*)d_in[IB[b][9]];
    blk[b].w1      = (const float*)d_in[IB[b][10]];
    blk[b].b1      = (const float*)d_in[IB[b][11]];
    blk[b].w2      = (const float*)d_in[IB[b][12]];
    blk[b].b2      = (const float*)d_in[IB[b][13]];
  }
  const float* x1 = (const float*)d_in[IX[0]];
  const float* x2 = (const float*)d_in[IX[1]];
  const float* x3 = (const float*)d_in[IX[2]];
  const float* x4 = (const float*)d_in[IX[3]];
  const float* x5 = (const float*)d_in[IX[4]];
  const float* route = (const float*)d_in[IROUTE];
  const int*   task  = (const int*)d_in[ITASK];
  const float* up1 = (const float*)d_in[IUP[0]];
  const float* up2 = (const float*)d_in[IUP[1]];
  const float* up3 = (const float*)d_in[IUP[2]];
  const float* up4 = (const float*)d_in[IUP[3]];

  float* tA = wsf + OFF_TA;
  float* tB = wsf + OFF_TB;
  float* dout_aux = dout + (out_size - 2);

  // zero the two aux-loss accumulators (output image is fully overwritten)
  zero_kernel<<<1, 32, 0, stream>>>(dout_aux, 2);

  // block1: d=256 @ 32x32
  transpose_cn_kernel<<<(256 * 1024 + 255) / 256, 256, 0, stream>>>(x5, tA, 256, 1024);
  run_block(tA, blk[0], 256, 32, 32, route, task, dout_aux, wsf, stream);
  {
    long tot = (long)64 * 64 * 128;
    upsample_kernel<1><<<(int)((tot + 255) / 256), 256, 0, stream>>>(
        tA, up1, x4, tB, 256, 32, 32);
  }
  // block2: d=128 @ 64x64
  run_block(tB, blk[1], 128, 64, 64, route, task, dout_aux, wsf, stream);
  {
    long tot = (long)128 * 128 * 64;
    upsample_kernel<1><<<(int)((tot + 255) / 256), 256, 0, stream>>>(
        tB, up2, x3, tA, 128, 64, 64);
  }
  // block3: d=64 @ 128x128
  run_block(tA, blk[2], 64, 128, 128, route, task, dout_aux, wsf, stream);
  {
    long tot = (long)256 * 256 * 32;
    upsample_kernel<1><<<(int)((tot + 255) / 256), 256, 0, stream>>>(
        tA, up3, x2, tB, 64, 128, 128);
  }
  // block4: d=32 @ 256x256
  run_block(tB, blk[3], 32, 256, 256, route, task, dout_aux, wsf, stream);
  {
    long tot = (long)512 * 512 * 16;
    upsample_kernel<0><<<(int)((tot + 255) / 256), 256, 0, stream>>>(
        tB, up4, x1, dout, 32, 256, 256);
  }
}
